// LocalCluster_14740327760103
// MI455X (gfx1250) — matrix operand order repaired and checked by reading, not compiled, not run on hardware
//
#include <hip/hip_runtime.h>
#include <hip/hip_bf16.h>
#include <cstdint>

typedef __attribute__((ext_vector_type(16))) _Float16 v16h;
typedef __attribute__((ext_vector_type(8)))  float    v8f;
typedef __attribute__((ext_vector_type(4)))  unsigned v4u;
typedef __attribute__((ext_vector_type(8)))  int      v8i;
typedef __attribute__((ext_vector_type(4)))  int      v4i;

#define EPSV 1e-12f

// Problem sizes (fixed by the reference)
#define NB    16          // batch
#define CIN   256         // in_d (= K of both GEMMs)
#define HWP   16384       // 128*128 pixels per image
#define OPROJ 512         // proj output channels
#define OMRG  256         // merge output channels
#define MCL   512         // m = n*FC*FH*FW
#define LLOC  4096        // L = 64*64
#define SCEN  64          // s = CS*CS

static __device__ __forceinline__ v8f vzero8() {
    v8f z;
#pragma unroll
    for (int i = 0; i < 8; ++i) z[i] = 0.0f;
    return z;
}

// ISA: generic LDS addresses carry the LDS byte offset in the low 32 bits.
static __device__ __forceinline__ unsigned lds_addr_of(const void* p) {
    return (unsigned)(uintptr_t)p;
}

// Tensor Data Mover: async 2D tile load global -> LDS.
// D# per CDNA5 ISA: group0 {count, lds_addr,
// global_addr, type=2}; group1 {data_size=2B, tensor dims (huge -> no OOB
// clip), tile_dim0/1, tensor_dim0_stride}. Remaining groups zero (2D tensor).
// This toolchain exposes the 6-arg builtin (clang-23 / therock-10.0).
static __device__ __forceinline__ void tdm_load_2d(unsigned ldsOff,
                                                   const _Float16* gptr,
                                                   unsigned tileD0,
                                                   unsigned tileD1,
                                                   unsigned long long strideEl) {
    unsigned long long ga = (unsigned long long)(uintptr_t)gptr;
    v4u g0;
    g0[0] = 1u;                                        // count=1, user mode
    g0[1] = ldsOff;                                    // lds_addr [63:32]
    g0[2] = (unsigned)(ga & 0xFFFFFFFFu);              // global_addr lo
    g0[3] = (unsigned)((ga >> 32) & 0x1FFFFFFu) | (2u << 30);  // hi | type=2
    const unsigned td0 = 0x7FFFFFFFu, td1 = 0x7FFFFFFFu;
    v8i g1;
    g1[0] = (int)(1u << 16);                           // data_size=1 (2 bytes)
    g1[1] = (int)((td0 & 0xFFFFu) << 16);              // tensor_dim0[15:0]@48
    g1[2] = (int)((td0 >> 16) | ((td1 & 0xFFFFu) << 16));  // d0 hi | d1 lo
    g1[3] = (int)((td1 >> 16) | (tileD0 << 16));       // d1 hi | tile_dim0
    g1[4] = (int)(tileD1 & 0xFFFFu);                   // tile_dim1 (dim2=0)
    g1[5] = (int)(unsigned)(strideEl & 0xFFFFFFFFull); // dim0_stride lo32
    g1[6] = (int)(unsigned)((strideEl >> 32) & 0xFFFFull);  // dim0_stride hi16
    g1[7] = 0;
    v4i z4;
    z4[0] = 0; z4[1] = 0; z4[2] = 0; z4[3] = 0;
    v8i z8;
#pragma unroll
    for (int i = 0; i < 8; ++i) z8[i] = 0;
    __builtin_amdgcn_tensor_load_to_lds(g0, g1, z4, z4, z8, 0);
}

// ---------------------------------------------------------------- converts
__global__ void k_f32_to_f16(const float* __restrict__ src,
                             _Float16* __restrict__ dst, int nElem) {
    int i = blockIdx.x * blockDim.x + threadIdx.x;
    if (i < nElem) dst[i] = (_Float16)src[i];
}

// ---------------------------------------------------------------- GEMM
// C(n) = A(M x 256) @ B_n(256 x 16384) + bias,  f16 inputs, f32 accum (WMMA).
// Double-buffered: A tiles via TDM (tensor_load_to_lds), B tiles via
// cooperative load + transpose-store, overlap with WMMA compute.
// mode 0: proj -> scatter into y2[(mi*4096+l)*64 + c2] (fused pixel-shuffle)
// mode 1: merge -> direct store out[((n*256)+o)*16384 + p]
__global__ void k_gemm(const _Float16* __restrict__ A,
                       const _Float16* __restrict__ Ball,
                       const float* __restrict__ bias,
                       float* __restrict__ outF,
                       int mode) {
    __shared__ __align__(32) _Float16 As[2][128 * 32];  // [row][k]
    __shared__ __align__(32) _Float16 Bs[2][128 * 32];  // [col][k] (K-major)
    const int n  = blockIdx.z;
    const int mB = blockIdx.y * 128;
    const int pB = blockIdx.x * 128;
    const _Float16* __restrict__ B = Ball + (size_t)n * CIN * HWP;

    const int t    = threadIdx.x;
    const int lane = t & 31, wid = t >> 5;
    const int wm   = wid & 3, wn = wid >> 2;     // 4 (M) x 2 (N) waves
    const int lr   = lane & 15, hl = lane >> 4;

    // cooperative B tile: 32(K) x 128(N) -> LDS transposed [N][K]
    auto loadB = [&](int bi, int kk) {
#pragma unroll
        for (int i = 0; i < 2; ++i) {
            int q  = t + i * 256;
            int kr = q >> 4, ch = q & 15;
            uint4 v = *(const uint4*)&B[(size_t)(kk + kr) * HWP + pB + ch * 8];
            __align__(16) _Float16 tmp[8];
            *(uint4*)tmp = v;
#pragma unroll
            for (int j = 0; j < 8; ++j) Bs[bi][(ch * 8 + j) * 32 + kr] = tmp[j];
        }
    };

    v8f acc[2][4];
#pragma unroll
    for (int mf = 0; mf < 2; ++mf)
#pragma unroll
        for (int nf = 0; nf < 4; ++nf) acc[mf][nf] = vzero8();

    // Prologue: stage tile 0
    if (wid == 0)
        tdm_load_2d(lds_addr_of(&As[0][0]), &A[(size_t)mB * CIN], 32, 128, CIN);
    loadB(0, 0);
    if (wid == 0) __builtin_amdgcn_s_wait_tensorcnt(0);
    __syncthreads();

    const int NSTEP = CIN / 32;    // 8
    for (int step = 0; step < NSTEP; ++step) {
        const int bi = step & 1;
        // Prefetch next tile into the other buffer (async TDM + coop B)
        if (step + 1 < NSTEP) {
            if (wid == 0)
                tdm_load_2d(lds_addr_of(&As[bi ^ 1][0]),
                            &A[(size_t)mB * CIN + (step + 1) * 32], 32, 128,
                            CIN);
            loadB(bi ^ 1, (step + 1) * 32);
        }

        // A fragments: lane holds M=lr, K = e + 16*hl (matches B)
        v16h afrag[2];
#pragma unroll
        for (int mf = 0; mf < 2; ++mf) {
            int row = wm * 32 + mf * 16 + lr;
#pragma unroll
            for (int e = 0; e < 16; ++e) {
                int K = e + 16 * hl;
                afrag[mf][e] = As[bi][row * 32 + K];
            }
        }
#pragma unroll
        for (int nf = 0; nf < 4; ++nf) {
            int col = wn * 64 + nf * 16 + lr;
            // B layout: lane holds N=lr, K = e + 16*hl -> contiguous 32B
            v16h bfrag = *(const v16h*)&Bs[bi][col * 32 + hl * 16];
#pragma unroll
            for (int mf = 0; mf < 2; ++mf)
                acc[mf][nf] = __builtin_amdgcn_wmma_f32_16x16x32_f16(
                    false, afrag[mf], false, bfrag, (short)0, acc[mf][nf],
                    false, false);
        }
        if (step + 1 < NSTEP) {
            if (wid == 0) __builtin_amdgcn_s_wait_tensorcnt(0);
            __syncthreads();
        }
    }

    // Epilogue: C layout — VGPR r: M = r + 8*hl, N = lr
#pragma unroll
    for (int mf = 0; mf < 2; ++mf) {
#pragma unroll
        for (int nf = 0; nf < 4; ++nf) {
#pragma unroll
            for (int r = 0; r < 8; ++r) {
                int o = mB + wm * 32 + mf * 16 + r + 8 * hl;
                int p = pB + wn * 64 + nf * 16 + lr;
                float v = acc[mf][nf][r] + bias[o];
                if (mode == 0) {
                    // y(n,o,H,W) -> y2[mi][l][c2] pixel-shuffle scatter
                    int fc = o >> 6, c2 = o & 63;
                    int Hi = p >> 7, Wi = p & 127;
                    int mi = (n << 5) | (fc << 2) | ((Hi >> 6) << 1) | (Wi >> 6);
                    int l  = ((Hi & 63) << 6) | (Wi & 63);
                    outF[(size_t)(mi * LLOC + l) * 64 + c2] = v;
                } else {
                    outF[((size_t)n * OMRG + o) * HWP + p] = v;
                }
            }
        }
    }
}

// ---------------------------------------------------------------- centers
__global__ void k_center(const float* __restrict__ y2,
                         _Float16* __restrict__ ncbf,
                         float* __restrict__ acc33) {
    __shared__ float cen[64];
    __shared__ float nrm;
    int mi = blockIdx.x >> 6, si = blockIdx.x & 63;
    int sh = si >> 3, sw = si & 7;
    int c  = threadIdx.x;
    float sum = 0.0f;
    for (int a = 0; a < 8; ++a)
        for (int b = 0; b < 8; ++b) {
            int l = ((sh * 8 + a) << 6) | (sw * 8 + b);
            sum += y2[(size_t)(mi * LLOC + l) * 64 + c];
        }
    float mean = sum * (1.0f / 64.0f);
    cen[c] = mean;
    __syncthreads();
    if (c == 0) {
        float s2 = 0.0f;
        for (int j = 0; j < 32; ++j) s2 += cen[j] * cen[j];
        nrm = fmaxf(sqrtf(s2), EPSV);
    }
    __syncthreads();
    int g = mi * 64 + si;
    if (c < 32) {
        ncbf[(size_t)g * 32 + c] = (_Float16)(mean / nrm);
    } else {
        acc33[(size_t)g * 33 + (c - 32)] = mean;
        if (c == 63) acc33[(size_t)g * 33 + 32] = 1.0f;
    }
}

// ---------------------------------------------------------------- normalize x_point
__global__ void k_normx(const float* __restrict__ y2,
                        _Float16* __restrict__ nxbf) {
    int t   = blockIdx.x * 256 + threadIdx.x;
    int row = t >> 5, c = t & 31;        // one (mi,l) per 32 lanes
    float v = y2[(size_t)row * 64 + c];
    float s = v * v;
    for (int d = 16; d > 0; d >>= 1) s += __shfl_xor(s, d, 32);
    nxbf[(size_t)row * 32 + c] = (_Float16)(v / fmaxf(sqrtf(s), EPSV));
}

// ---------------------------------------------------------------- sim + argmax + segment atomics
__global__ void k_sim(const _Float16* __restrict__ nxbf,
                      const _Float16* __restrict__ ncbf,
                      const float* __restrict__ y2,
                      const float* __restrict__ alphaP,
                      const float* __restrict__ betaP,
                      float* __restrict__ acc33,
                      float* __restrict__ vals,
                      int* __restrict__ idxArr) {
    __shared__ __align__(32) _Float16 ncs[64 * 32];
    const int mi = blockIdx.y;
    const int rB = blockIdx.x * 128;
    const int t  = threadIdx.x;
    *(uint4*)&ncs[t * 8] = *(const uint4*)&ncbf[(size_t)mi * 2048 + t * 8];
    __syncthreads();

    const int lane = t & 31, wid = t >> 5;
    const int lr = lane & 15, hl = lane >> 4;
    const int rowT = rB + wid * 16;

    v16h a;
    {
        const _Float16* __restrict__ src =
            &nxbf[(size_t)(mi * LLOC + rowT + lr) * 32];
#pragma unroll
        for (int e = 0; e < 16; ++e) {
            int K = e + 16 * hl;
            a[e] = src[K];
        }
    }
    v8f acc[4];
#pragma unroll
    for (int nf = 0; nf < 4; ++nf) acc[nf] = vzero8();
#pragma unroll
    for (int nf = 0; nf < 4; ++nf) {
        int col = nf * 16 + lr;
        v16h b = *(const v16h*)&ncs[col * 32 + hl * 16];
        acc[nf] = __builtin_amdgcn_wmma_f32_16x16x32_f16(
            false, a, false, b, (short)0, acc[nf], false, false);
    }

    const float alpha = alphaP[0], beta = betaP[0];
#pragma unroll
    for (int r = 0; r < 8; ++r) {
        float mv = -3.4e38f;
        int   mj = 0;
#pragma unroll
        for (int nf = 0; nf < 4; ++nf) {
            float v = acc[nf][r];
            int  nn = nf * 16 + lr;
            if (v > mv) { mv = v; mj = nn; }
        }
        for (int d = 1; d < 16; d <<= 1) {
            float ov = __shfl_xor(mv, d, 32);
            int   oj = __shfl_xor(mj, d, 32);
            if (ov > mv || (ov == mv && oj < mj)) { mv = ov; mj = oj; }
        }
        int row = rowT + r + 8 * hl;
        float sv = 1.0f / (1.0f + __expf(-(alpha * mv + beta)));
        size_t rl = (size_t)mi * LLOC + row;
        if (lr == 0) { vals[rl] = sv; idxArr[rl] = mj; }
        size_t g = (size_t)mi * 64 + mj;
        const float* __restrict__ xv = &y2[rl * 64 + 32];   // value half
        atomicAdd(&acc33[g * 33 + lr],      sv * xv[lr]);
        atomicAdd(&acc33[g * 33 + 16 + lr], sv * xv[16 + lr]);
        if (lr == 0) atomicAdd(&acc33[g * 33 + 32], sv);
    }
}

// ---------------------------------------------------------------- finalize centers
__global__ void k_final(const float* __restrict__ acc33,
                        float* __restrict__ newc) {
    int t = blockIdx.x * 256 + threadIdx.x;   // MCL*SCEN*32
    int g = t >> 5, c = t & 31;
    newc[t] = acc33[(size_t)g * 33 + c] / acc33[(size_t)g * 33 + 32];
}

// ---------------------------------------------------------------- gather new_x
__global__ void k_newx(const float* __restrict__ vals,
                       const int* __restrict__ idxArr,
                       const float* __restrict__ newc,
                       _Float16* __restrict__ nxout) {
    size_t t = (size_t)blockIdx.x * 256 + threadIdx.x;  // 16*256*16384
    int p  = (int)(t & 16383);
    int ch = (int)((t >> 14) & 255);
    int n  = (int)(t >> 22);
    int fc = ch >> 5, c = ch & 31;
    int Hi = p >> 7, Wi = p & 127;
    int mi = (n << 5) | (fc << 2) | ((Hi >> 6) << 1) | (Wi >> 6);
    int l  = ((Hi & 63) << 6) | (Wi & 63);
    size_t rl = (size_t)mi * LLOC + l;
    size_t g  = (size_t)mi * 64 + (size_t)idxArr[rl];
    nxout[t]  = (_Float16)(vals[rl] * newc[g * 32 + c]);
}

// ---------------------------------------------------------------- host
extern "C" void kernel_launch(void* const* d_in, const int* in_sizes, int n_in,
                              void* d_out, int out_size, void* d_ws, size_t ws_size,
                              hipStream_t stream) {
    (void)in_sizes; (void)n_in; (void)out_size; (void)ws_size;
    const float* x     = (const float*)d_in[0];
    const float* Wp    = (const float*)d_in[1];
    const float* bp    = (const float*)d_in[2];
    const float* Wm    = (const float*)d_in[3];
    const float* bm    = (const float*)d_in[4];
    const float* alpha = (const float*)d_in[5];
    const float* beta  = (const float*)d_in[6];

    char* w = (char*)d_ws;
    auto alloc = [&](size_t bytes) -> char* {
        char* p = w;
        w += (bytes + 255) & ~(size_t)255;
        return p;
    };
    const size_t nX   = (size_t)NB * CIN * HWP;        // 67,108,864
    _Float16* xbf    = (_Float16*)alloc(nX * 2);
    _Float16* Wpbf   = (_Float16*)alloc((size_t)OPROJ * CIN * 2);
    _Float16* Wmbf   = (_Float16*)alloc((size_t)OMRG * CIN * 2);
    float*    y2     = (float*)alloc((size_t)MCL * LLOC * 64 * 4);  // 537 MB
    _Float16* ncbf   = (_Float16*)alloc((size_t)MCL * SCEN * 32 * 2);
    _Float16* nxbf   = (_Float16*)alloc((size_t)MCL * LLOC * 32 * 2);
    float*    acc33  = (float*)alloc((size_t)MCL * SCEN * 33 * 4);
    float*    vals   = (float*)alloc((size_t)MCL * LLOC * 4);
    int*      idxArr = (int*)alloc((size_t)MCL * LLOC * 4);
    float*    newc   = (float*)alloc((size_t)MCL * SCEN * 32 * 4);
    _Float16* newxbf = (_Float16*)alloc(nX * 2);

    k_f32_to_f16<<<(int)(nX / 256), 256, 0, stream>>>(x, xbf, (int)nX);
    k_f32_to_f16<<<OPROJ * CIN / 256, 256, 0, stream>>>(Wp, Wpbf, OPROJ * CIN);
    k_f32_to_f16<<<OMRG * CIN / 256, 256, 0, stream>>>(Wm, Wmbf, OMRG * CIN);
    k_gemm<<<dim3(HWP / 128, OPROJ / 128, NB), 256, 0, stream>>>(
        Wpbf, xbf, bp, y2, 0);
    k_center<<<MCL * SCEN, 64, 0, stream>>>(y2, ncbf, acc33);
    k_normx<<<(MCL * LLOC * 32) / 256, 256, 0, stream>>>(y2, nxbf);
    k_sim<<<dim3(LLOC / 128, MCL), 256, 0, stream>>>(
        nxbf, ncbf, y2, alpha, beta, acc33, vals, idxArr);
    k_final<<<(MCL * SCEN * 32) / 256, 256, 0, stream>>>(acc33, newc);
    k_newx<<<(int)(nX / 256), 256, 0, stream>>>(vals, idxArr, newc, newxbf);
    k_gemm<<<dim3(HWP / 128, OMRG / 128, NB), 256, 0, stream>>>(
        Wmbf, newxbf, bm, (float*)d_out, 1);
}
